// RNN_79937931313742
// MI455X (gfx1250) — matrix operand order repaired and checked by reading, not compiled, not run on hardware
//
#include <hip/hip_runtime.h>
#include <hip/hip_bf16.h>
#include <hip/hip_fp16.h>

typedef __attribute__((ext_vector_type(2)))  float     v2f;
typedef __attribute__((ext_vector_type(8)))  float     v8f;
typedef __attribute__((ext_vector_type(4)))  float     v4f;
typedef __attribute__((ext_vector_type(8)))  _Float16  v8h;
typedef __attribute__((ext_vector_type(16))) _Float16  v16h;

#define SEQ    128
#define BATCH  64
#define HID    1024
#define EMB    512
#define VOCAB  10000

// ---------------------------------------------------------------------------
// Recurrence step: h_new = tanh( x @ Wx^T + h_prev @ Wh^T + b )
// x is BATCH x Kx row-major, or gathered from emb via tok (layer 0).
// One wave = one 16x16 tile with V_WMMA_F32_16X16X4_F32 (full fp32: this path
// feeds back through tanh for 128 steps, so no precision downgrade here).
// Two interleaved accumulators break the serial RAW chain on the C operand.
// Tiles: 4 (M over BATCH) x 64 (N over HID) = 256; 64 blocks x 4 waves.
// ---------------------------------------------------------------------------
__global__ void rnn_step_kernel(const float* __restrict__ x,
                                const int*   __restrict__ tok,
                                const float* __restrict__ emb,
                                int Kx,
                                const float* __restrict__ Wx,     // HID x Kx
                                const float* __restrict__ Wh,     // HID x HID
                                const float* __restrict__ bias,   // HID
                                const float* __restrict__ h_prev, // BATCH x HID
                                float*       __restrict__ h_new,  // BATCH x HID
                                _Float16*    __restrict__ h_new_f16)
{
    const int lane  = threadIdx.x & 31;
    const int wave  = threadIdx.x >> 5;
    const int tile  = blockIdx.x * (blockDim.x >> 5) + wave;
    const int mbase = (tile & 3)  * 16;   // over BATCH
    const int nbase = (tile >> 2) * 16;   // over HID

    const int ml   = lane & 15;           // M index for A-frag, N for B/C-frag
    const int koff = (lane >> 4) * 2;     // K sub-offset (f32 16x16x4 layout)

    const float* arow = tok ? (emb + (size_t)tok[mbase + ml] * Kx)
                            : (x   + (size_t)(mbase + ml) * Kx);
    const float* bxrow = Wx + (size_t)(nbase + ml) * Kx;   // B[k][n] = Wx[n][k]

    v8f acc0 = {}, acc1 = {};
    for (int k = 0; k < Kx; k += 8) {
        v2f a0 = *(const v2f*)(arow  + k + koff);
        v2f b0 = *(const v2f*)(bxrow + k + koff);
        v2f a1 = *(const v2f*)(arow  + k + 4 + koff);
        v2f b1 = *(const v2f*)(bxrow + k + 4 + koff);
        acc0 = __builtin_amdgcn_wmma_f32_16x16x4_f32(
                   false, a0, false, b0, (short)0, acc0, false, false);
        acc1 = __builtin_amdgcn_wmma_f32_16x16x4_f32(
                   false, a1, false, b1, (short)0, acc1, false, false);
    }

    const float* hrow  = h_prev + (size_t)(mbase + ml) * HID;
    const float* bhrow = Wh     + (size_t)(nbase + ml) * HID;
    for (int k = 0; k < HID; k += 8) {
        v2f a0 = *(const v2f*)(hrow  + k + koff);
        v2f b0 = *(const v2f*)(bhrow + k + koff);
        v2f a1 = *(const v2f*)(hrow  + k + 4 + koff);
        v2f b1 = *(const v2f*)(bhrow + k + 4 + koff);
        acc0 = __builtin_amdgcn_wmma_f32_16x16x4_f32(
                   false, a0, false, b0, (short)0, acc0, false, false);
        acc1 = __builtin_amdgcn_wmma_f32_16x16x4_f32(
                   false, a1, false, b1, (short)0, acc1, false, false);
    }

    // C/D layout: VGPR r -> M = r (lanes 0-15) or 8+r (lanes 16-31), N = lane&15
    const float bn    = bias[nbase + ml];
    const int   mrow0 = mbase + ((lane >> 4) << 3);
#pragma unroll
    for (int r = 0; r < 8; ++r) {
        const int m = mrow0 + r;
        const float v = tanhf(acc0[r] + acc1[r] + bn);
        h_new[(size_t)m * HID + nbase + ml] = v;
        if (h_new_f16)
            h_new_f16[(size_t)m * HID + nbase + ml] = (_Float16)v;
    }
}

// ---------------------------------------------------------------------------
// fp32 -> fp16 conversion (one shot for Wout), vectorized 4-wide
// ---------------------------------------------------------------------------
__global__ void f32_to_f16_kernel(const float* __restrict__ in,
                                  _Float16* __restrict__ out, int n4)
{
    int i = blockIdx.x * blockDim.x + threadIdx.x;
    const int stride = gridDim.x * blockDim.x;
    for (; i < n4; i += stride) {
        const v4f v = *(const v4f*)(in + 4 * (size_t)i);
        _Float16* o = out + 4 * (size_t)i;
        o[0] = (_Float16)v.x; o[1] = (_Float16)v.y;
        o[2] = (_Float16)v.z; o[3] = (_Float16)v.w;
    }
}

// ---------------------------------------------------------------------------
// Output projection over ALL timesteps at once:
//   logits[(t*B + b), v] = h1f16[t*B + b, :] . Wout16[v, :] + bout[v]
// (8192 x 10000 x 1024) with V_WMMA_F32_16X16X32_F16 (f32 accumulate).
// Register blocking: one wave = 32(M) x 80(N) outputs = 2 A-frags x 5 B-frags
// = 10 WMMAs per K-step of 32 (~23 FLOP per byte of L2 traffic).
// B fragments are 32 contiguous bytes per lane -> loaded as a single v16h so
// the b128 pair lands directly in the WMMA source octet (no packing VALU).
// Wave-jobs are M-major so a block's 8 waves share the same Wout fragments.
// ---------------------------------------------------------------------------
#define MB 2   // 16-row M tiles per wave
#define NB 5   // 16-col N tiles per wave

union h16frag { v16h v; v8h h[2]; };

__global__ void logits_kernel(const _Float16* __restrict__ A,    // (SEQ*BATCH) x HID
                              const _Float16* __restrict__ Wf,   // VOCAB x HID
                              const float*    __restrict__ bout, // VOCAB
                              float*          __restrict__ out)  // (SEQ*BATCH) x VOCAB
{
    const int NGM = (SEQ * BATCH) / (16 * MB);   // 256
    const int NGN = VOCAB / (16 * NB);           // 125
    const int lane = threadIdx.x & 31;
    const int wave = threadIdx.x >> 5;
    const int job  = blockIdx.x * (blockDim.x >> 5) + wave;
    if (job >= NGM * NGN) return;                // wave-uniform, EXEC stays full

    const int mbase = (job % NGM) * (16 * MB);   // M-major: waves in a block
    const int nbase = (job / NGM) * (16 * NB);   //   share the same N group

    const int nl = lane & 15;
    const int hi = lane >> 4;
    const int koffA = hi * 16;   // A frag, contiguous k order used here: lane-half picks K 0..15/16..31
    const int koffB = hi * 16;   // B frag, same contiguous k order as A: lane-half picks K 0..15/16..31

    const _Float16* arow[MB];
    const _Float16* brow[NB];
#pragma unroll
    for (int mi = 0; mi < MB; ++mi)
        arow[mi] = A + (size_t)(mbase + mi * 16 + nl) * HID;
#pragma unroll
    for (int ni = 0; ni < NB; ++ni)
        brow[ni] = Wf + (size_t)(nbase + ni * 16 + nl) * HID;

    v8f acc[MB][NB];
#pragma unroll
    for (int mi = 0; mi < MB; ++mi)
#pragma unroll
        for (int ni = 0; ni < NB; ++ni)
            acc[mi][ni] = (v8f){};

    for (int k = 0; k < HID; k += 32) {
        h16frag afrag[MB];
        v16h    bfrag[NB];
#pragma unroll
        for (int mi = 0; mi < MB; ++mi) {
            afrag[mi].h[0] = *(const v8h*)(arow[mi] + k + koffA);
            afrag[mi].h[1] = *(const v8h*)(arow[mi] + k + 8 + koffA);
            __builtin_prefetch(arow[mi] + k + 64, 0, 0);  // next K-slab (A streams)
        }
#pragma unroll
        for (int ni = 0; ni < NB; ++ni)
            bfrag[ni] = *(const v16h*)(brow[ni] + k + koffB);  // 32B contiguous
#pragma unroll
        for (int mi = 0; mi < MB; ++mi)
#pragma unroll
            for (int ni = 0; ni < NB; ++ni)
                acc[mi][ni] = __builtin_amdgcn_wmma_f32_16x16x32_f16(
                    false, afrag[mi].v, false, bfrag[ni], (short)0, acc[mi][ni],
                    false, false);
    }

#pragma unroll
    for (int mi = 0; mi < MB; ++mi) {
        const int mrow0 = mbase + mi * 16 + hi * 8;
#pragma unroll
        for (int ni = 0; ni < NB; ++ni) {
            const int col = nbase + ni * 16 + nl;
            const float bn = bout[col];
#pragma unroll
            for (int r = 0; r < 8; ++r)
                out[(size_t)(mrow0 + r) * VOCAB + col] = acc[mi][ni][r] + bn;
        }
    }
}

// ---------------------------------------------------------------------------
extern "C" void kernel_launch(void* const* d_in, const int* in_sizes, int n_in,
                              void* d_out, int out_size, void* d_ws, size_t ws_size,
                              hipStream_t stream)
{
    const int*   inputs = (const int*)  d_in[0];   // SEQ x BATCH
    const float* hidden = (const float*)d_in[1];   // 2 x BATCH x HID
    const float* emb    = (const float*)d_in[2];   // VOCAB x EMB
    const float* W0x    = (const float*)d_in[3];   // HID x EMB
    const float* W0h    = (const float*)d_in[4];   // HID x HID
    const float* b0     = (const float*)d_in[5];
    const float* W1x    = (const float*)d_in[6];   // HID x HID
    const float* W1h    = (const float*)d_in[7];   // HID x HID
    const float* b1     = (const float*)d_in[8];
    const float* Wout   = (const float*)d_in[9];   // VOCAB x HID
    const float* bout   = (const float*)d_in[10];

    float* logits       = (float*)d_out;                        // SEQ*BATCH*VOCAB
    float* final_hidden = logits + (size_t)SEQ * BATCH * VOCAB; // 2*BATCH*HID

    // workspace layout (~38.3 MB total)
    char* ws = (char*)d_ws;
    _Float16* Wout16 = (_Float16*)ws;                                   // VOCAB*HID f16
    _Float16* h1f16  = (_Float16*)(ws + (size_t)VOCAB * HID * 2);       // SEQ*BATCH*HID f16
    float*    hbuf   = (float*)(ws + (size_t)VOCAB * HID * 2
                                   + (size_t)SEQ * BATCH * HID * 2);    // 4 x BATCH*HID f32
    float* h0pp[2] = { hbuf,                   hbuf + 1 * BATCH * HID };
    float* h1pp[2] = { hbuf + 2 * BATCH * HID, hbuf + 3 * BATCH * HID };

    // initial hidden state -> ping-pong buffers
    hipMemcpyAsync(h0pp[0], hidden,               (size_t)BATCH * HID * sizeof(float),
                   hipMemcpyDeviceToDevice, stream);
    hipMemcpyAsync(h1pp[0], hidden + BATCH * HID, (size_t)BATCH * HID * sizeof(float),
                   hipMemcpyDeviceToDevice, stream);

    // one-time Wout fp32 -> fp16
    f32_to_f16_kernel<<<2048, 256, 0, stream>>>(Wout, Wout16, (VOCAB * HID) / 4);

    // sequential recurrence (critical path), full fp32 WMMA
    int cur = 0;
    for (int t = 0; t < SEQ; ++t) {
        const int nxt = cur ^ 1;
        rnn_step_kernel<<<64, 128, 0, stream>>>(
            nullptr, inputs + t * BATCH, emb, EMB,
            W0x, W0h, b0, h0pp[cur], h0pp[nxt], (_Float16*)nullptr);
        rnn_step_kernel<<<64, 128, 0, stream>>>(
            h0pp[nxt], nullptr, nullptr, HID,
            W1x, W1h, b1, h1pp[cur], h1pp[nxt],
            h1f16 + (size_t)t * BATCH * HID);
        cur = nxt;
    }

    // final hidden state -> tail of d_out
    hipMemcpyAsync(final_hidden,               h0pp[cur],
                   (size_t)BATCH * HID * sizeof(float), hipMemcpyDeviceToDevice, stream);
    hipMemcpyAsync(final_hidden + BATCH * HID, h1pp[cur],
                   (size_t)BATCH * HID * sizeof(float), hipMemcpyDeviceToDevice, stream);

    // big mixed-precision output projection over all timesteps
    const int njobs  = ((SEQ * BATCH) / (16 * MB)) * (VOCAB / (16 * NB)); // 32000
    const int blocks = (njobs + 7) / 8;                                   // 8 waves/block
    logits_kernel<<<blocks, 256, 0, stream>>>(h1f16, Wout16, bout, logits);
}
